// QuadTreeDecoder_50062138802243
// MI455X (gfx1250) — matrix operand order repaired and checked by reading, not compiled, not run on hardware
//
#include <hip/hip_runtime.h>
#include <hip/hip_bf16.h>

typedef __attribute__((ext_vector_type(16))) _Float16 v16h;
typedef __attribute__((ext_vector_type(8)))  float    v8f;

#define BOND  64
#define RANK  16
#define BATCH 128

// A-operand (16x32 f16) element->K map (per lane half = lane>>4):
// lane<16:  e0..7 -> K0..7,   e8..15 -> K16..23
// lane>=16: e0..7 -> K8..15,  e8..15 -> K24..31
__device__ __forceinline__ int amap(int e, int half) {
    return (e < 8) ? (half * 8 + e) : (16 + half * 8 + (e - 8));
}

// A-layout position permutation within a 32-wide K block: swap octets 1 and 2.
// Stored this way, lane `half` reads its 16 A elements as ONE contiguous run
// at [half*16, half*16+16)  ->  two ds_load_b128, zero selects, zero branches.
__device__ __forceinline__ int permA(int k) {          // k in [0,32)
    return (k >= 8 && k < 24) ? (k ^ 24) : k;
}

__device__ __forceinline__ v16h ld16h(const _Float16* p) {
    return *reinterpret_cast<const v16h*>(p);
}

__device__ __forceinline__ v8f wmma_f16(v16h a, v16h b, v8f c) {
    return __builtin_amdgcn_wmma_f32_16x16x32_f16(false, a, false, b, (short)0, c,
                                                  false, false);
}

// ---------------------------------------------------------------------------
// Generic quadtree level: one workgroup per node, 8 waves, wave w = batch tile.
//   lat = x_n(128x64) * fi_n^T(64x16), *= scale
//   child_q = lat * f_q(16x64) + x_n   -> scattered to quadtree child layout
// x layout: (node, batch, 64) node-major so each node block is contiguous.
// LDS layouts are WMMA-operand-ready f16 with zero padding (no conditionals).
// ---------------------------------------------------------------------------
__global__ __launch_bounds__(256)
void qt_level_kernel(const float* __restrict__ xin,
                     const float* __restrict__ f_in,
                     const float* __restrict__ f_tl,
                     const float* __restrict__ f_tr,
                     const float* __restrict__ f_bl,
                     const float* __restrict__ f_br,
                     const float* __restrict__ scale,
                     float* __restrict__ xout,
                     int off, int gridw)
{
    // fi rows [r][k]        : B operand of GEMM1 (plain, contiguous reads)
    // fq^T rows [o][32]     : B operand of GEMM2; cols 16..31 zero (K pad)
    // lat  [w][row][32]     : A operand of GEMM2, plain K layout, k>=16 zero
    __shared__ __align__(32) _Float16 lds_fi[RANK * BOND];        //  2 KB
    __shared__ __align__(32) _Float16 lds_fq[4 * BOND * 32];      // 16 KB
    __shared__ __align__(32) _Float16 lds_lat[8 * 16 * 32];       //  8 KB
    __shared__ float lds_scale[RANK];

    const int tid  = threadIdx.x;
    const int nd   = blockIdx.x;
    const int base = (off + nd) * RANK * BOND;

    // stage factors (f32 -> f16, quadrants transposed to [o][r])
    for (int i = tid; i < RANK * BOND; i += 256) {
        int r = i >> 6, o = i & 63;
        int p = o * 32 + r;
        lds_fi[i]            = (_Float16)f_in[base + i];
        lds_fq[0 * 2048 + p] = (_Float16)f_tl[base + i];
        lds_fq[1 * 2048 + p] = (_Float16)f_tr[base + i];
        lds_fq[2 * 2048 + p] = (_Float16)f_bl[base + i];
        lds_fq[3 * 2048 + p] = (_Float16)f_br[base + i];
    }
    // zero the K>=16 pad of fq^T rows (1024 halfs per quadrant, as dwords)
    {
        unsigned* z = (unsigned*)lds_fq;
        for (int i = tid; i < 2048; i += 256) {
            int q = i >> 9, rem = i & 511, o = rem >> 3, u = rem & 7;
            z[q * 1024 + o * 16 + 8 + u] = 0u;
        }
    }
    // zero the K>=16 pad of lat (positions 16..31, as dwords)
    {
        unsigned* z = (unsigned*)lds_lat;
        for (int i = tid; i < 1024; i += 256) {
            int wr = i >> 3, u = i & 7;
            z[wr * 16 + 8 + u] = 0u;
        }
    }
    if (tid < RANK) lds_scale[tid] = scale[(off + nd) * RANK + tid];
    __syncthreads();

    const int lane = tid & 31;
    const int w    = tid >> 5;          // wave id == batch tile
    const int half = lane >> 4;
    const int ln   = lane & 15;

    const float* xnode = xin + (size_t)nd * BATCH * BOND;
    const int    row   = w * 16 + ln;   // batch row for A operand

    // ---- GEMM1: lat(16x16) = x_tile(16x64) * fi^T(64x16), f32 accumulate ----
    v8f lat = {};
#pragma unroll
    for (int kb = 0; kb < BOND; kb += 32) {
        v16h a;
#pragma unroll
        for (int e = 0; e < 16; ++e)
            a[e] = (_Float16)xnode[row * BOND + kb + half * 16 + e];
        v16h b = ld16h(lds_fi + ln * BOND + kb + half * 16);
        lat = wmma_f16(a, b, lat);
    }

    // scale, convert, stage into plain K layout (real K<16 positions only)
    {
        float s = lds_scale[ln];
        int pos = ln;   // plain K, ln<16
#pragma unroll
        for (int j = 0; j < 8; ++j)
            lds_lat[w * 512 + (j + 8 * half) * 32 + pos] = (_Float16)(lat[j] * s);
    }

    // A operand for GEMM2 (shared by all quadrants): one contiguous v16h load
    v16h a2 = ld16h(lds_lat + w * 512 + ln * 32 + half * 16);

    const int gy = nd / gridw, gx = nd % gridw;

#pragma unroll
    for (int q = 0; q < 4; ++q) {
        const _Float16* fqt = lds_fq + q * 2048;
        const int n2 = (2 * gy + (q >> 1)) * (2 * gridw) + (2 * gx + (q & 1));
        float* xchild = xout + (size_t)n2 * BATCH * BOND;
#pragma unroll
        for (int nt = 0; nt < 4; ++nt) {
            v16h b2 = ld16h(fqt + (nt * 16 + ln) * 32 + half * 16);
            v8f c;
#pragma unroll
            for (int j = 0; j < 8; ++j)             // residual: C = x_n (f32!)
                c[j] = xnode[(w * 16 + j + 8 * half) * BOND + nt * 16 + ln];
            c = wmma_f16(a2, b2, c);
#pragma unroll
            for (int j = 0; j < 8; ++j)
                xchild[(w * 16 + j + 8 * half) * BOND + nt * 16 + ln] = c[j];
        }
    }
}

// ---------------------------------------------------------------------------
// Last level fused with the head: children never hit HBM (saves ~268 MB).
// child tile staged in LDS in plain K f16 layout, head operands are pure
// v16h LDS loads, then pixel-shuffle scatter to (B,3,256,256).
// ---------------------------------------------------------------------------
__global__ __launch_bounds__(256)
void qt_last_head_kernel(const float* __restrict__ xin,
                         const float* __restrict__ f_in,
                         const float* __restrict__ f_tl,
                         const float* __restrict__ f_tr,
                         const float* __restrict__ f_bl,
                         const float* __restrict__ f_br,
                         const float* __restrict__ scale,
                         const float* __restrict__ head_w,
                         const float* __restrict__ head_b,
                         float* __restrict__ out,
                         int off, int gridw)
{
    __shared__ __align__(32) _Float16 lds_fi[RANK * BOND];        //  2 KB
    __shared__ __align__(32) _Float16 lds_fq[4 * BOND * 32];      // 16 KB
    __shared__ __align__(32) _Float16 lds_lat[8 * 16 * 32];       //  8 KB
    __shared__ __align__(32) _Float16 lds_child[8 * 16 * BOND];   // 16 KB
    __shared__ __align__(32) _Float16 lds_hw[48 * BOND];          //  6 KB
    __shared__ float lds_scale[RANK];
    __shared__ float lds_hb[48];

    const int tid  = threadIdx.x;
    const int nd   = blockIdx.x;
    const int base = (off + nd) * RANK * BOND;

    for (int i = tid; i < RANK * BOND; i += 256) {
        int r = i >> 6, o = i & 63;
        int p = o * 32 + r;
        lds_fi[i]            = (_Float16)f_in[base + i];
        lds_fq[0 * 2048 + p] = (_Float16)f_tl[base + i];
        lds_fq[1 * 2048 + p] = (_Float16)f_tr[base + i];
        lds_fq[2 * 2048 + p] = (_Float16)f_bl[base + i];
        lds_fq[3 * 2048 + p] = (_Float16)f_br[base + i];
    }
    {
        unsigned* z = (unsigned*)lds_fq;
        for (int i = tid; i < 2048; i += 256) {
            int q = i >> 9, rem = i & 511, o = rem >> 3, u = rem & 7;
            z[q * 1024 + o * 16 + 8 + u] = 0u;
        }
    }
    {
        unsigned* z = (unsigned*)lds_lat;
        for (int i = tid; i < 1024; i += 256) {
            int wr = i >> 3, u = i & 7;
            z[wr * 16 + 8 + u] = 0u;
        }
    }
    for (int i = tid; i < 48 * BOND; i += 256) lds_hw[i] = (_Float16)head_w[i];
    if (tid < RANK) lds_scale[tid] = scale[(off + nd) * RANK + tid];
    if (tid < 48)   lds_hb[tid]    = head_b[tid];
    __syncthreads();

    const int lane = tid & 31;
    const int w    = tid >> 5;
    const int half = lane >> 4;
    const int ln   = lane & 15;

    const float* xnode = xin + (size_t)nd * BATCH * BOND;
    const int    row   = w * 16 + ln;

    // ---- GEMM1 ----
    v8f lat = {};
#pragma unroll
    for (int kb = 0; kb < BOND; kb += 32) {
        v16h a;
#pragma unroll
        for (int e = 0; e < 16; ++e)
            a[e] = (_Float16)xnode[row * BOND + kb + half * 16 + e];
        v16h b = ld16h(lds_fi + ln * BOND + kb + half * 16);
        lat = wmma_f16(a, b, lat);
    }
    {
        float s = lds_scale[ln];
        int pos = ln;
#pragma unroll
        for (int j = 0; j < 8; ++j)
            lds_lat[w * 512 + (j + 8 * half) * 32 + pos] = (_Float16)(lat[j] * s);
    }
    v16h a2 = ld16h(lds_lat + w * 512 + ln * 32 + half * 16);

    const int gy = nd / gridw, gx = nd % gridw;
    _Float16* childbuf = lds_child + w * 16 * BOND;

#pragma unroll 1
    for (int q = 0; q < 4; ++q) {
        const _Float16* fqt = lds_fq + q * 2048;
        const int gy2 = 2 * gy + (q >> 1);
        const int gx2 = 2 * gx + (q & 1);

        // ---- GEMM2 -> stage child tile in LDS (plain K f16, K=64 full) ----
#pragma unroll
        for (int nt = 0; nt < 4; ++nt) {
            v16h b2 = ld16h(fqt + (nt * 16 + ln) * 32 + half * 16);
            v8f c;
#pragma unroll
            for (int j = 0; j < 8; ++j)
                c[j] = xnode[(w * 16 + j + 8 * half) * BOND + nt * 16 + ln];
            c = wmma_f16(a2, b2, c);
            const int k   = nt * 16 + ln;
            const int pos = k;
#pragma unroll
            for (int j = 0; j < 8; ++j)
                childbuf[(j + 8 * half) * BOND + pos] = (_Float16)c[j];
        }

        // ---- head: y(16x48) = child(16x64) * head_w^T + head_b ----
#pragma unroll
        for (int nt = 0; nt < 3; ++nt) {
            const int o_out = nt * 16 + ln;
            v8f c;
            float bias = lds_hb[o_out];
#pragma unroll
            for (int j = 0; j < 8; ++j) c[j] = bias;
#pragma unroll
            for (int kb = 0; kb < BOND; kb += 32) {
                v16h a3 = ld16h(childbuf + ln * BOND + kb + half * 16);
                v16h b3 = ld16h(lds_hw + o_out * BOND + kb + half * 16);
                c = wmma_f16(a3, b3, c);
            }
            // pixel-shuffle scatter: o = py*12 + px*3 + ch
            const int py = o_out / 12;
            const int px = (o_out % 12) / 3;
            const int ch = o_out % 3;
            const int rowp = gy2 * 4 + py;
            const int colp = gx2 * 4 + px;
#pragma unroll
            for (int j = 0; j < 8; ++j) {
                int bb = w * 16 + j + 8 * half;
                out[(((size_t)bb * 3 + ch) * 256 + rowp) * 256 + colp] = c[j];
            }
        }
    }
}

extern "C" void kernel_launch(void* const* d_in, const int* in_sizes, int n_in,
                              void* d_out, int out_size, void* d_ws, size_t ws_size,
                              hipStream_t stream) {
    const float* x   = (const float*)d_in[0];   // (128,1,64)
    const float* fin = (const float*)d_in[1];
    const float* ftl = (const float*)d_in[2];
    const float* ftr = (const float*)d_in[3];
    const float* fbl = (const float*)d_in[4];
    const float* fbr = (const float*)d_in[5];
    const float* sc  = (const float*)d_in[6];
    const float* hw  = (const float*)d_in[7];
    const float* hb  = (const float*)d_in[8];
    float* out = (float*)d_out;
    float* ws  = (float*)d_ws;

    // node-major activation buffers (node, 128, 64); levels 1..5 only
    const size_t blk = (size_t)BATCH * BOND;
    float* x1 = ws;                  // 4 nodes
    float* x2 = x1 + 4    * blk;     // 16
    float* x3 = x2 + 16   * blk;     // 64
    float* x4 = x3 + 64   * blk;     // 256
    float* x5 = x4 + 256  * blk;     // 1024  (total ~44.7 MB of ws)

    qt_level_kernel<<<1,    256, 0, stream>>>(x,  fin, ftl, ftr, fbl, fbr, sc, x1,   0,  1);
    qt_level_kernel<<<4,    256, 0, stream>>>(x1, fin, ftl, ftr, fbl, fbr, sc, x2,   1,  2);
    qt_level_kernel<<<16,   256, 0, stream>>>(x2, fin, ftl, ftr, fbl, fbr, sc, x3,   5,  4);
    qt_level_kernel<<<64,   256, 0, stream>>>(x3, fin, ftl, ftr, fbl, fbr, sc, x4,  21,  8);
    qt_level_kernel<<<256,  256, 0, stream>>>(x4, fin, ftl, ftr, fbl, fbr, sc, x5,  85, 16);
    qt_last_head_kernel<<<1024, 256, 0, stream>>>(x5, fin, ftl, ftr, fbl, fbr, sc,
                                                  hw, hb, out, 341, 32);
}
